// NeuralRDE_84507776516481
// MI455X (gfx1250) — compile-verified, matrix operand order checked by reading, not run on hardware
//
#include <hip/hip_runtime.h>
#include <hip/hip_bf16.h>

// ---------------- problem constants ----------------
#define Bn     512
#define Hn     64
#define Ln     528
#define Dn     32
#define Wn     128
#define NIn    64
#define NSTEPS 256
#define BH     (Bn * Hn)
#define LS1    ((NIn + 1) * (Ln + 1))   // logsig stride over b  = 65*529
#define LS2    (Ln + 1)                 // logsig stride over ni = 529

typedef __bf16 bf16_t;
typedef bf16_t v16bf __attribute__((ext_vector_type(16)));
typedef float  v8f   __attribute__((ext_vector_type(8)));
typedef unsigned int v4u __attribute__((ext_vector_type(4)));
typedef int v8i __attribute__((ext_vector_type(8)));
typedef int v4i __attribute__((ext_vector_type(4)));

// ---------------- small device helpers ----------------
__device__ __forceinline__ unsigned short f2bf(float f) {
  unsigned int u = __float_as_uint(f);
  u += 0x7FFFu + ((u >> 16) & 1u);          // round-to-nearest-even
  return (unsigned short)(u >> 16);
}

__device__ __forceinline__ float fast_tanh(float x) {
  float xc = fminf(fmaxf(x, -15.f), 15.f);
  float e  = __builtin_amdgcn_exp2f(xc * 2.88539008177792681f);  // exp(2x)
  return (e - 1.f) * __builtin_amdgcn_rcpf(e + 1.f);
}

__device__ __forceinline__ float fast_silu(float x) {
  float a = fminf(-x * 1.44269504088896341f, 126.f);
  float e = __builtin_amdgcn_exp2f(a);                            // exp(-x)
  return x * __builtin_amdgcn_rcpf(1.f + e);
}

// ---------------- fp32 -> bf16 weight conversion ----------------
__global__ void kCvt(const float* __restrict__ src, unsigned short* __restrict__ dst, int n) {
  int i = blockIdx.x * 256 + threadIdx.x;
  if (i < n) dst[i] = f2bf(src[i]);
}

// ---------------- y0 = x0 @ l1_w.T + l1_b ----------------
__global__ void kInitY(const float* __restrict__ x0, const float* __restrict__ l1w,
                       const float* __restrict__ l1b, float* __restrict__ y) {
  int i = blockIdx.x * 256 + threadIdx.x;   // 0..BH-1
  int b = i >> 6, h = i & 63;
  float s = l1b[h];
#pragma unroll
  for (int d = 0; d < Dn; ++d) s += x0[b * Dn + d] * l1w[h * Dn + d];
  y[i] = s;
}

// ---------------- one MLP layer: 16 x K  @  (128 x K)^T, silu ----------------
template <int K, bool TOGLOBAL>
__device__ __forceinline__ void mlp_layer(
    const unsigned short* __restrict__ a_src,   // LDS: 16 rows x K bf16
    const unsigned short* __restrict__ Wbf,     // global: 128 x K bf16 (row-major, out x in)
    const float* __restrict__ bias,             // 128
    unsigned short* __restrict__ out_lds,       // LDS: 16 x 128 bf16
    unsigned short* __restrict__ gout,          // global out if TOGLOBAL
    int rb, int lane, int wave) {
  const int jlo = lane & 15;
  const int jhi = lane >> 4;
  constexpr int KC = K / 32;
  v16bf afrag[KC];
#pragma unroll
  for (int kc = 0; kc < KC; ++kc)
    afrag[kc] = *(const v16bf*)((const bf16_t*)a_src + jlo * K + kc * 32 + jhi * 16);
#pragma unroll
  for (int nt = 0; nt < 2; ++nt) {
    const int ntile = wave + nt * 4;            // 4 waves cover 8 col-tiles (N=128)
    const int n     = ntile * 16 + jlo;
    const bf16_t* bp = (const bf16_t*)Wbf + (size_t)n * K + jhi * 16;
    v8f c = {};
#pragma unroll
    for (int kc = 0; kc < KC; ++kc) {
      v16bf bfrag = *(const v16bf*)(bp + kc * 32);
      c = __builtin_amdgcn_wmma_f32_16x16x32_bf16(false, afrag[kc], false, bfrag,
                                                  (short)0, c, false, false);
    }
    const float bv = bias[n];
#pragma unroll
    for (int i = 0; i < 8; ++i) {
      const int row = i + jhi * 8;              // C layout: reg i, lane-half jhi
      unsigned short u = f2bf(fast_silu(c[i] + bv));
      if (TOGLOBAL) gout[(size_t)(rb + row) * Wn + n] = u;
      else          out_lds[row * Wn + n] = u;
    }
  }
}

// ---------------- stage prep: y-combine + 3-layer MLP -> h3 (bf16) ----------------
__global__ __launch_bounds__(128) void kStagePrep(
    float* __restrict__ y, const float* __restrict__ kst,  // k1..k6 contiguous, stride BH
    const float* __restrict__ ts,
    const unsigned short* __restrict__ W0bf, const float* __restrict__ b0,
    const unsigned short* __restrict__ W1bf, const float* __restrict__ b1,
    const unsigned short* __restrict__ W2bf, const float* __restrict__ b2,
    unsigned short* __restrict__ h3bf,
    float a1, float a2, float a3, float a4, float a5, int nk, int combine) {
  __shared__ __align__(32) unsigned short a0_lds[16 * Hn];
  __shared__ __align__(32) unsigned short h1_lds[16 * Wn];
  __shared__ __align__(32) unsigned short h2_lds[16 * Wn];

  const int tid  = threadIdx.x;
  const int lane = tid & 31;
  const int wave = tid >> 5;
  const int rb   = blockIdx.x * 16;
  const float dt = (ts[128] - ts[0]) * (1.0f / NSTEPS);

  for (int e = tid; e < 16 * Hn; e += 128) {
    const int r = e >> 6, c = e & 63;
    const size_t gi = (size_t)(rb + r) * Hn + c;
    float yv = y[gi];
    if (combine) {  // finish previous RK step: y += dt * sum(Bi * ki)
      yv += dt * (0.09646076681806523f   * kst[gi]
                + 0.01f                  * kst[BH + gi]
                + 0.4798896504144996f    * kst[2 * BH + gi]
                + 1.379008574103742f     * kst[3 * BH + gi]
                + (-3.290069515436081f)  * kst[4 * BH + gi]
                + 2.324710524099774f     * kst[5 * BH + gi]);
      y[gi] = yv;
    }
    float s = 0.f;
    if (nk > 0) s += a1 * kst[gi];
    if (nk > 1) s += a2 * kst[BH + gi];
    if (nk > 2) s += a3 * kst[2 * BH + gi];
    if (nk > 3) s += a4 * kst[3 * BH + gi];
    if (nk > 4) s += a5 * kst[4 * BH + gi];
    a0_lds[r * Hn + c] = f2bf(yv + dt * s);
  }
  __syncthreads();
  mlp_layer<Hn, false>(a0_lds, W0bf, b0, h1_lds, nullptr, rb, lane, wave);
  __syncthreads();
  mlp_layer<Wn, false>(h1_lds, W1bf, b1, h2_lds, nullptr, rb, lane, wave);
  __syncthreads();
  mlp_layer<Wn, true>(h2_lds, W2bf, b2, nullptr, h3bf, rb, lane, wave);
}

// ---------------- big fused GEMM + tanh + sig-reduction -> k_s ----------------
// grid (B/16, H/4), 128 threads (4 waves); wave w handles h = by*4 + w.
__global__ __launch_bounds__(128) void kBigDot(
    const unsigned short* __restrict__ W3bf,   // (H*L) x 128 bf16
    const float* __restrict__ b3,              // H*L
    const unsigned short* __restrict__ h3bf,   // B x 128 bf16
    const float* __restrict__ logsig, const float* __restrict__ intervals,
    const float* __restrict__ ts, float* __restrict__ kout, int step, float cs) {
  __shared__ float sig_lds[16][Ln];
  __shared__ __align__(32) unsigned short a_lds[16 * Wn];

  const int tid  = threadIdx.x;
  const int lane = tid & 31;
  const int wave = tid >> 5;
  const int rb   = blockIdx.x * 16;
  const int h    = blockIdx.y * 4 + wave;

  // t and searchsorted(intervals, t) -- uniform scalar work
  const float t0 = ts[0];
  const float dt = (ts[128] - t0) * (1.0f / NSTEPS);
  const float t  = t0 + ((float)step + cs) * dt;
  int idx = 0;
  while (idx < NIn && intervals[idx] < t) idx++;

  // ---- stage sig tile (16 x 528 f32, row stride LS1) into LDS ----
  const size_t gofs = (size_t)rb * LS1 + (size_t)idx * LS2 + 1;
#if __has_builtin(__builtin_amdgcn_tensor_load_to_lds) && \
    __has_builtin(__builtin_amdgcn_s_wait_tensorcnt)
  // Tensor Data Mover: one async 2-D tile DMA issued by wave 0 only
  // (TENSOR ops issue once per wave and ignore EXEC).
  if (wave == 0) {
    const unsigned long long ga = (unsigned long long)(const void*)(logsig + gofs);
    const unsigned int lds_off  = (unsigned int)(unsigned long long)(void*)&sig_lds[0][0];
    v4u g0;
    g0[0] = 1u;                                   // count=1, load, user descriptor
    g0[1] = lds_off;                              // D#.lds_addr (bytes)
    g0[2] = (unsigned int)ga;                     // D#.global_addr[31:0]
    g0[3] = (unsigned int)((ga >> 32) & 0x1FFFFFFull) | (2u << 30);  // addr[56:32], type=2
    const unsigned int TD = 0x7FFFFFFFu;          // huge tensor dims: no OOB clamp
    v8i g1;
    g1[0] = (int)(2u << 16);                      // wg_mask=0, data_size=4B
    g1[1] = (int)((TD & 0xFFFFu) << 16);          // abar=0, tensor_dim0.lo16
    g1[2] = (int)((TD >> 16) | ((TD & 0xFFFFu) << 16));   // td0.hi16, td1.lo16
    g1[3] = (int)((TD >> 16) | ((unsigned)Ln << 16));     // td1.hi16, tile_dim0=528
    g1[4] = 16;                                   // tile_dim1=16, tile_dim2=0
    g1[5] = (int)LS1;                             // tensor_dim0_stride low 32 (=34385)
    g1[6] = 0;                                    // stride0.hi16, stride1.lo16
    g1[7] = 0;                                    // stride1.hi32
    v4i gz  = {0, 0, 0, 0};
    v8i gz8 = {0, 0, 0, 0, 0, 0, 0, 0};
    __builtin_amdgcn_tensor_load_to_lds(g0, g1, gz, gz, gz8, 0);
  }
#else
  for (int e = tid; e < 16 * Ln; e += 128) {
    const int r = e / Ln, l = e - r * Ln;
    sig_lds[r][l] = logsig[gofs + (size_t)r * LS1 + l];
  }
#endif

  // A tile (16 x 128 bf16) into LDS cooperatively (overlaps with the TDM DMA)
  {
    const unsigned int* src = (const unsigned int*)(h3bf + (size_t)rb * Wn);
    unsigned int* dst = (unsigned int*)a_lds;
    for (int e = tid; e < 16 * Wn / 2; e += 128) dst[e] = src[e];
  }
#if __has_builtin(__builtin_amdgcn_tensor_load_to_lds) && \
    __has_builtin(__builtin_amdgcn_s_wait_tensorcnt)
  if (wave == 0) __builtin_amdgcn_s_wait_tensorcnt(0);
#endif
  __syncthreads();

  const int jlo = lane & 15;
  const int jhi = lane >> 4;
  v16bf afrag[4];  // A fragments reused across all 33 column tiles
#pragma unroll
  for (int kc = 0; kc < 4; ++kc)
    afrag[kc] = *(const v16bf*)((const bf16_t*)a_lds + jlo * Wn + kc * 32 + jhi * 16);

  const size_t nbase = (size_t)h * Ln;
  float acc[8];
#pragma unroll
  for (int i = 0; i < 8; ++i) acc[i] = 0.f;

#pragma unroll 2   // two independent WMMA chains in flight
  for (int ct = 0; ct < Ln / 16; ++ct) {
    const int n = ct * 16 + jlo;
    const bf16_t* bp = (const bf16_t*)W3bf + (nbase + n) * Wn + jhi * 16;
    if (ct + 1 < Ln / 16) __builtin_prefetch((const void*)(bp + 16 * Wn), 0, 0);
    v8f c = {};
#pragma unroll
    for (int kc = 0; kc < 4; ++kc) {
      v16bf bfrag = *(const v16bf*)(bp + kc * 32);
      c = __builtin_amdgcn_wmma_f32_16x16x32_bf16(false, afrag[kc], false, bfrag,
                                                  (short)0, c, false, false);
    }
    const float bias = b3[nbase + n];
#pragma unroll
    for (int i = 0; i < 8; ++i) {
      const int row = i + jhi * 8;
      acc[i] += fast_tanh(c[i] + bias) * sig_lds[row][n];
    }
  }
  // reduce over the 16 columns held across each lane-half
#pragma unroll
  for (int i = 0; i < 8; ++i) {
    float a = acc[i];
    a += __shfl_xor(a, 1, 32);
    a += __shfl_xor(a, 2, 32);
    a += __shfl_xor(a, 4, 32);
    a += __shfl_xor(a, 8, 32);
    acc[i] = a;
  }
  if (jlo == 0) {
#pragma unroll
    for (int i = 0; i < 8; ++i)
      kout[(size_t)(rb + i + jhi * 8) * Hn + h] = acc[i];
  }
}

// ---------------- final combine + head + softmax ----------------
__global__ __launch_bounds__(64) void kFinal(
    const float* __restrict__ y, const float* __restrict__ kst,
    const float* __restrict__ ts, const float* __restrict__ l2w,
    const float* __restrict__ l2b, float* __restrict__ out) {
  __shared__ float yf[Hn];
  __shared__ float logits[10];
  const int b = blockIdx.x, hh = threadIdx.x;
  const float dt = (ts[128] - ts[0]) * (1.0f / NSTEPS);
  const size_t gi = (size_t)b * Hn + hh;
  float v = y[gi] + dt * (0.09646076681806523f   * kst[gi]
                        + 0.01f                  * kst[BH + gi]
                        + 0.4798896504144996f    * kst[2 * BH + gi]
                        + 1.379008574103742f     * kst[3 * BH + gi]
                        + (-3.290069515436081f)  * kst[4 * BH + gi]
                        + 2.324710524099774f     * kst[5 * BH + gi]);
  yf[hh] = v;
  __syncthreads();
  if (hh < 10) {
    float s = l2b[hh];
#pragma unroll
    for (int j = 0; j < Hn; ++j) s += yf[j] * l2w[hh * Hn + j];
    logits[hh] = s;
  }
  __syncthreads();
  if (hh == 0) {
    float m = logits[0];
    for (int c = 1; c < 10; ++c) m = fmaxf(m, logits[c]);
    float e[10], sum = 0.f;
#pragma unroll
    for (int c = 0; c < 10; ++c) {
      e[c] = __builtin_amdgcn_exp2f((logits[c] - m) * 1.44269504088896341f);
      sum += e[c];
    }
    const float r = 1.f / sum;
#pragma unroll
    for (int c = 0; c < 10; ++c) out[b * 10 + c] = e[c] * r;
  }
}

// ---------------- host driver ----------------
extern "C" void kernel_launch(void* const* d_in, const int* in_sizes, int n_in,
                              void* d_out, int out_size, void* d_ws, size_t ws_size,
                              hipStream_t stream) {
  const float* ts        = (const float*)d_in[0];
  const float* logsig    = (const float*)d_in[1];
  const float* x0        = (const float*)d_in[2];
  const float* intervals = (const float*)d_in[3];
  const float* w0 = (const float*)d_in[4];  const float* b0 = (const float*)d_in[5];
  const float* w1 = (const float*)d_in[6];  const float* b1 = (const float*)d_in[7];
  const float* w2 = (const float*)d_in[8];  const float* b2 = (const float*)d_in[9];
  const float* w3 = (const float*)d_in[10]; const float* b3 = (const float*)d_in[11];
  const float* l1w = (const float*)d_in[12]; const float* l1b = (const float*)d_in[13];
  const float* l2w = (const float*)d_in[14]; const float* l2b = (const float*)d_in[15];

  char* ws = (char*)d_ws;
  unsigned short* W0bf = (unsigned short*)(ws + 0);          //  128*64  bf16
  unsigned short* W1bf = (unsigned short*)(ws + 16384);      //  128*128 bf16
  unsigned short* W2bf = (unsigned short*)(ws + 49152);      //  128*128 bf16
  unsigned short* W3bf = (unsigned short*)(ws + 81920);      //  33792*128 bf16
  unsigned short* h3bf = (unsigned short*)(ws + 8732672);    //  B*128 bf16
  float*          y    = (float*)(ws + 8863744);             //  B*H f32
  float*          kst  = (float*)(ws + 8994816);             //  6 * B*H f32

  kCvt<<<(Wn * Hn + 255) / 256, 256, 0, stream>>>(w0, W0bf, Wn * Hn);
  kCvt<<<(Wn * Wn + 255) / 256, 256, 0, stream>>>(w1, W1bf, Wn * Wn);
  kCvt<<<(Wn * Wn + 255) / 256, 256, 0, stream>>>(w2, W2bf, Wn * Wn);
  kCvt<<<(Hn * Ln * Wn + 255) / 256, 256, 0, stream>>>(w3, W3bf, Hn * Ln * Wn);
  kInitY<<<BH / 256, 256, 0, stream>>>(x0, l1w, l1b, y);

  static const float Atab[6][5] = {
      {0.f, 0.f, 0.f, 0.f, 0.f},
      {0.161f, 0.f, 0.f, 0.f, 0.f},
      {-0.008480655492356989f, 0.335480655492357f, 0.f, 0.f, 0.f},
      {2.8971530571054935f, -6.359448489975075f, 4.3622954328695815f, 0.f, 0.f},
      {5.325864828439257f, -11.748883564062828f, 7.4955393428898365f,
       -0.09249506636175525f, 0.f},
      {5.86145544294642f, -12.92096931784711f, 8.159367898576159f,
       -0.071584973281401f, -0.028269050394068383f}};
  static const float cstab[6] = {0.f, 0.161f, 0.327f, 0.9f, 0.9800255409045097f, 1.0f};

  const dim3 gA(Bn / 16), bA(128);
  const dim3 gB(Bn / 16, Hn / 4), bB(128);
  for (int i = 0; i < NSTEPS; ++i) {
    for (int s = 0; s < 6; ++s) {
      kStagePrep<<<gA, bA, 0, stream>>>(y, kst, ts, W0bf, b0, W1bf, b1, W2bf, b2, h3bf,
                                        Atab[s][0], Atab[s][1], Atab[s][2], Atab[s][3],
                                        Atab[s][4], s, (s == 0 && i > 0) ? 1 : 0);
      kBigDot<<<gB, bB, 0, stream>>>(W3bf, b3, h3bf, logsig, intervals, ts,
                                     kst + (size_t)s * BH, i, cstab[s]);
    }
  }
  kFinal<<<Bn, 64, 0, stream>>>(y, kst, ts, l2w, l2b, (float*)d_out);
}
